// SpatialAttention_24206435680951
// MI455X (gfx1250) — matrix operand order repaired and checked by reading, not compiled, not run on hardware
//
#include <hip/hip_runtime.h>
#include <hip/hip_bf16.h>
#include <math.h>

// ---------------------------------------------------------------------------
// SpatialAttention for MI455X (gfx1250): bf16 WMMA, fused two-pass softmax,
// TDM (tensor_load_to_lds) staging of the K operand in the fused pass.
// B=4, C=256, H=W=64, N=4096.
// ---------------------------------------------------------------------------

typedef __bf16 bf16_t;
typedef __attribute__((ext_vector_type(16))) __bf16 v16bf;
typedef __attribute__((ext_vector_type(8)))  __bf16 v8bf;
typedef __attribute__((ext_vector_type(8)))  float  v8f;

#define BB 4
#define CC 256
#define NN 4096

// ---- WMMA fragment helpers (CDNA5 16x16x32 bf16 layouts, ISA 7.12.2) ------

// A (16x32, MxK) from row-major f32 source with leading dim `ld`.
__device__ __forceinline__ v16bf a_frag_f32(const float* __restrict__ base,
                                            int ld, int lane) {
  const int m = lane & 15, hf = lane >> 4;
  const float* p0 = base + (size_t)m * ld + hf * 16;
  const float* p1 = p0 + 8;
  v16bf a;
#pragma unroll
  for (int i = 0; i < 8; ++i) {
    a[i]     = (bf16_t)p0[i];
    a[i + 8] = (bf16_t)p1[i];
  }
  return a;
}

// A (16x32) from row-major bf16 source: two 16B vector loads.
__device__ __forceinline__ v16bf a_frag_bf16(const bf16_t* __restrict__ base,
                                             int ld, int lane) {
  const int m = lane & 15, hf = lane >> 4;
  const bf16_t* p0 = base + (size_t)m * ld + hf * 16;
  union { v16bf v; v8bf h2[2]; } u;
  u.h2[0] = *(const v8bf*)(p0);
  u.h2[1] = *(const v8bf*)(p0 + 8);
  return u.v;
}

// B (32x16, KxN) from a K-contiguous ("transposed", [N][K]) bf16 source.
__device__ __forceinline__ v16bf b_frag_bf16(const bf16_t* __restrict__ baseT,
                                             int ld, int lane) {
  const int n = lane & 15, hf = lane >> 4;
  const bf16_t* p = baseT + (size_t)n * ld + hf * 16;
  union { v16bf v; v8bf h2[2]; } u;
  u.h2[0] = *(const v8bf*)(p);
  u.h2[1] = *(const v8bf*)(p + 8);
  return u.v;
}

__device__ __forceinline__ v8f wmma_bf16(v16bf a, v16bf b, v8f c) {
  return __builtin_amdgcn_wmma_f32_16x16x32_bf16(
      false, a, false, b, (short)0, c, false, false);
}

// ---- TDM: DMA a [16 x 256] bf16 tile of Kt[N][C] into LDS with padding ----
// Row pitch in LDS becomes 512B + 16B pad = 528B (264 bf16) -> bank spread
// and still 16B-aligned for the v8bf fragment loads.
#define KB_LD 264

#if __has_builtin(__builtin_amdgcn_tensor_load_to_lds)
#define HAVE_TDM 1
typedef unsigned int u32x4 __attribute__((ext_vector_type(4)));
typedef int          i32x8 __attribute__((ext_vector_type(8)));
typedef int          i32x4 __attribute__((ext_vector_type(4)));

__device__ __forceinline__ unsigned lds_addr_of(const void* p) {
  return (unsigned)(uintptr_t)(const __attribute__((address_space(3))) void*)p;
}

__device__ __forceinline__ void tdm_load_k16(const bf16_t* gsrc, unsigned ldsoff) {
  const unsigned long long ga = (unsigned long long)(uintptr_t)gsrc;
  u32x4 g0;
  g0[0] = 1u;                                   // count=1, user descriptor
  g0[1] = ldsoff;                               // lds_addr (bytes)
  g0[2] = (unsigned)(ga & 0xffffffffu);         // global_addr[31:0] @ bits 95:64
  g0[3] = (unsigned)(ga >> 32) | (2u << 30);    // global_addr[56:32] | type=2
  i32x8 g1;
  // data_size=2B (1), pad_enable, pad_interval=128 dwords (6), pad_amount=4 dw (3)
  g1[0] = (1 << 16) | (1 << 20) | (6 << 22) | (3 << 25);
  g1[1] = (int)(256u << 16);                    // tensor_dim0 = 256   (bits 63:48)
  g1[2] = (int)(4096u << 16);                   // tensor_dim1 = 4096  (bits 95:80)
  g1[3] = (int)(256u << 16);                    // tile_dim0  = 256    (bits 127:112)
  g1[4] = 16;                                   // tile_dim1  = 16
  g1[5] = 256;                                  // tensor_dim0_stride = 256 elems
  g1[6] = 0;
  g1[7] = 0;
  i32x4 z4 = {0, 0, 0, 0};
  i32x8 z8 = {0, 0, 0, 0, 0, 0, 0, 0};
  // 6-arg form (clang-23 / therock-10.0 headers): groups 2/3 + extra block = 0.
  __builtin_amdgcn_tensor_load_to_lds(g0, g1, z4, z4, z8, 0);
}
#endif

// ---------------------------------------------------------------------------
// Kernel 1: 1x1-conv projections. Per batch: Y = W(256x256) * X(256x4096)+b.
// x staged into LDS transposed+bf16 with 16-element XOR swizzle (64 KiB).
// Outputs: Kt (w1) [B][N][C], Qt (w2) [B][N][C], V (w3) [B][C][N], all bf16.
// ---------------------------------------------------------------------------
#define K1_NB 128

__global__ __launch_bounds__(256) void proj_kernel(
    const float* __restrict__ x,
    const float* __restrict__ w1, const float* __restrict__ b1,
    const float* __restrict__ w2, const float* __restrict__ b2,
    const float* __restrict__ w3, const float* __restrict__ b3,
    bf16_t* __restrict__ Kt, bf16_t* __restrict__ Qt, bf16_t* __restrict__ V) {
  __shared__ bf16_t xT[K1_NB * CC];  // (n,c) at n*CC + ((c>>4 ^ n&15)<<4) + (c&15)

  const int b  = blockIdx.x >> 5;
  const int nb = blockIdx.x & 31;
  const int n0 = nb * K1_NB;
  const int tid = threadIdx.x;
  const float* xb = x + (size_t)b * CC * NN;

#pragma unroll 1
  for (int it = 0; it < (CC * K1_NB / 4) / 256; ++it) {
    const int f  = it * 256 + tid;
    const int c  = f >> 5;
    const int nq = f & 31;
    const float4 v4 = *(const float4*)(xb + (size_t)c * NN + n0 + nq * 4);
    float vals[4] = {v4.x, v4.y, v4.z, v4.w};
#pragma unroll
    for (int j = 0; j < 4; ++j) {
      const int nl = nq * 4 + j;
      xT[nl * CC + ((((c >> 4) ^ (nl & 15)) << 4) | (c & 15))] = (bf16_t)vals[j];
    }
  }
  __syncthreads();

  const int wave = tid >> 5, lane = tid & 31;
  const int nt = wave * 16;
  const int jl = lane & 15, hf = lane >> 4;

  v16bf bfr[8];
#pragma unroll
  for (int kk = 0; kk < 8; ++kk) {
    const int row = nt + jl;
    const int kb  = kk * 2 + hf;
    const bf16_t* p = &xT[row * CC + ((kb ^ (row & 15)) << 4)];
    union { v16bf v; v8bf h2[2]; } u;
    u.h2[0] = *(const v8bf*)(p);
    u.h2[1] = *(const v8bf*)(p + 8);
    bfr[kk] = u.v;
  }

#pragma unroll
  for (int p = 0; p < 3; ++p) {
    const float* Wp = (p == 0) ? w1 : (p == 1) ? w2 : w3;
    const float* Bp = (p == 0) ? b1 : (p == 1) ? b2 : b3;
#pragma unroll 1
    for (int ot = 0; ot < 16; ++ot) {
      v8f acc = {};
#pragma unroll
      for (int kk = 0; kk < 8; ++kk) {
        v16bf af = a_frag_f32(Wp + (size_t)(ot * 16) * CC + kk * 32, CC, lane);
        acc = wmma_bf16(af, bfr[kk], acc);
      }
      const int nglob = n0 + nt + jl;
      if (p < 2) {
        bf16_t* dst = ((p == 0) ? Kt : Qt) + ((size_t)b * NN + nglob) * CC + ot * 16;
#pragma unroll
        for (int r = 0; r < 8; ++r)
          dst[r + 8 * hf] = (bf16_t)(acc[r] + Bp[ot * 16 + r + 8 * hf]);
      } else {
#pragma unroll
        for (int r = 0; r < 8; ++r) {
          const int o = ot * 16 + r + 8 * hf;
          V[((size_t)b * CC + o) * NN + nglob] = (bf16_t)(acc[r] + Bp[o]);
        }
      }
    }
  }
}

// ---------------------------------------------------------------------------
// Kernel 2: column softmax stats (softmax over i for each column j).
// One wave owns a 16-wide j-tile; online max/sum over all 4096 i via WMMA.
// ---------------------------------------------------------------------------
__global__ __launch_bounds__(256) void stats_kernel(
    const bf16_t* __restrict__ Qt, const bf16_t* __restrict__ Kt,
    float* __restrict__ colmax, float* __restrict__ colsum) {
  const int b  = blockIdx.x >> 5;
  const int jb = blockIdx.x & 31;
  const int wave = threadIdx.x >> 5, lane = threadIdx.x & 31;
  const int j0 = jb * 128 + wave * 16;

  const bf16_t* Qb = Qt + (size_t)b * NN * CC;
  const bf16_t* Kb = Kt + (size_t)b * NN * CC;

  v16bf bfr[8];
#pragma unroll
  for (int kk = 0; kk < 8; ++kk)
    bfr[kk] = b_frag_bf16(Kb + (size_t)j0 * CC + kk * 32, CC, lane);

  float mx = -1e30f, sm = 0.f;
#pragma unroll 1
  for (int i0 = 0; i0 < NN; i0 += 16) {
    if (i0 + 16 < NN)  // prefetch next Q i-tile (global_prefetch_b8)
      __builtin_prefetch(Qb + (size_t)(i0 + 16) * CC + (lane << 4), 0, 0);
    v8f acc = {};
#pragma unroll
    for (int kk = 0; kk < 8; ++kk) {
      v16bf af = a_frag_bf16(Qb + (size_t)i0 * CC + kk * 32, CC, lane);
      acc = wmma_bf16(af, bfr[kk], acc);
    }
    float tm = acc[0];
#pragma unroll
    for (int r = 1; r < 8; ++r) tm = fmaxf(tm, acc[r]);
    const float nm = fmaxf(mx, tm);
    sm *= __expf(mx - nm);
#pragma unroll
    for (int r = 0; r < 8; ++r) sm += __expf(acc[r] - nm);
    mx = nm;
  }
  const float mo = __shfl_xor(mx, 16, 32);
  const float so = __shfl_xor(sm, 16, 32);
  const float nm = fmaxf(mx, mo);
  const float s  = sm * __expf(mx - nm) + so * __expf(mo - nm);
  if (lane < 16) {
    colmax[(size_t)b * NN + j0 + lane] = nm;
    colsum[(size_t)b * NN + j0 + lane] = s;
  }
}

// ---------------------------------------------------------------------------
// Kernel 3: fused recompute + P.V^T + residual, with TDM-staged K tiles.
// Block owns (b, 16-wide m-tile). Wave (cg,ns): cg = half the channels,
// ns = 1024-wide n-chunk. Per 16-wide j-tile: TDM prefetches the next K tile
// while WMMA consumes the current one from LDS; every second tile feeds the
// 32-deep P.V^T WMMA step. ds_add_f32 reduction, then residual write.
// ---------------------------------------------------------------------------
#define P_LD 40

__global__ __launch_bounds__(256) void attnout_kernel(
    const float* __restrict__ x, const bf16_t* __restrict__ Qt,
    const bf16_t* __restrict__ Kt, const bf16_t* __restrict__ V,
    const float* __restrict__ colmax, const float* __restrict__ colsum,
    const float* __restrict__ alpha_p, float* __restrict__ out) {
  __shared__ float  ob[CC * 16];            // 16 KiB
  __shared__ bf16_t pb[8][16 * P_LD];       // 10 KiB
#if HAVE_TDM
  __shared__ bf16_t kb[8][2][16 * KB_LD];   // 132 KiB: per-wave double buffer
#endif

  const int b  = blockIdx.x >> 8;
  const int mt = blockIdx.x & 255;
  const int m0 = mt * 16;
  const int tid = threadIdx.x;
  const int wave = tid >> 5, lane = tid & 31;
  const int cg = wave & 1, ns = wave >> 1;
  const int jl = lane & 15, hf = lane >> 4;

  for (int i = tid; i < CC * 16; i += 256) ob[i] = 0.f;
  __syncthreads();

  const bf16_t* Qb = Qt + (size_t)b * NN * CC;
  const bf16_t* Kb = Kt + (size_t)b * NN * CC;
  const bf16_t* Vb = V + (size_t)b * CC * NN;

  v16bf qf[8];
#pragma unroll
  for (int kk = 0; kk < 8; ++kk)
    qf[kk] = a_frag_bf16(Qb + (size_t)m0 * CC + kk * 32, CC, lane);

  v8f zero = {};
  v8f oacc[8];
#pragma unroll
  for (int t = 0; t < 8; ++t) oacc[t] = zero;

  bf16_t* myP = pb[wave];
  const int nbeg = ns * (NN / 4);
  const int NT = (NN / 4) / 16;  // 64 j-tiles of 16 per wave

#if HAVE_TDM
  tdm_load_k16(Kb + (size_t)nbeg * CC, lds_addr_of(&kb[wave][0][0]));
#endif

#pragma unroll 1
  for (int jt = 0; jt < NT; ++jt) {
    const int j0 = nbeg + jt * 16;
#if HAVE_TDM
    if (jt + 1 < NT) {
      tdm_load_k16(Kb + (size_t)(j0 + 16) * CC,
                   lds_addr_of(&kb[wave][(jt + 1) & 1][0]));
      __builtin_amdgcn_s_wait_tensorcnt(1);
    } else {
      __builtin_amdgcn_s_wait_tensorcnt(0);
    }
    const bf16_t* ksrc = &kb[wave][jt & 1][0];
    const int kld = KB_LD;
#else
    const bf16_t* ksrc = Kb + (size_t)j0 * CC;
    const int kld = CC;
#endif
    // ---- logits tile: attn[m0..+15, j0..+15] ----------------------------
    v8f acc = zero;
#pragma unroll
    for (int kk = 0; kk < 8; ++kk)
      acc = wmma_bf16(qf[kk], b_frag_bf16(ksrc + kk * 32, kld, lane), acc);

    const int j = j0 + jl;
    const float cm  = colmax[(size_t)b * NN + j];
    const float inv = 1.0f / colsum[(size_t)b * NN + j];
#pragma unroll
    for (int r = 0; r < 8; ++r) {
      const float pv = __expf(acc[r] - cm) * inv;
      myP[(r + 8 * hf) * P_LD + (jt & 1) * 16 + jl] = (bf16_t)pv;
    }

    // ---- every 2nd tile: out[c, m0..] += V[c, n0..n0+31] * P^T ----------
    if (jt & 1) {
      const int n0 = j0 - 16;
      v16bf pf = b_frag_bf16(myP, P_LD, lane);
#pragma unroll
      for (int ct = 0; ct < 8; ++ct) {
        const int c0 = (cg * 8 + ct) * 16;
        v16bf vf = a_frag_bf16(Vb + (size_t)c0 * NN + n0, NN, lane);
        oacc[ct] = wmma_bf16(vf, pf, oacc[ct]);
      }
    }
  }

#pragma unroll
  for (int ct = 0; ct < 8; ++ct) {
    const int c0 = (cg * 8 + ct) * 16;
#pragma unroll
    for (int r = 0; r < 8; ++r)
      atomicAdd(&ob[(c0 + r + 8 * hf) * 16 + jl], oacc[ct][r]);
  }
  __syncthreads();

  const float alpha = alpha_p[0];
  for (int i = tid; i < CC * 16; i += 256) {
    const int c = i >> 4, m = i & 15;
    const size_t gi = ((size_t)b * CC + c) * NN + m0 + m;
    out[gi] = x[gi] + alpha * ob[i];
  }
}

// ---------------------------------------------------------------------------
// Launch. Inputs: x, w1, b1, w2, b2, w3, b3, alpha (all f32).
// Workspace: Kt/Qt/V bf16 (3 x 8 MiB) + colmax/colsum f32 ~25.3 MB.
// ---------------------------------------------------------------------------
extern "C" void kernel_launch(void* const* d_in, const int* in_sizes, int n_in,
                              void* d_out, int out_size, void* d_ws, size_t ws_size,
                              hipStream_t stream) {
  const float* x  = (const float*)d_in[0];
  const float* w1 = (const float*)d_in[1];
  const float* b1 = (const float*)d_in[2];
  const float* w2 = (const float*)d_in[3];
  const float* b2 = (const float*)d_in[4];
  const float* w3 = (const float*)d_in[5];
  const float* b3 = (const float*)d_in[6];
  const float* al = (const float*)d_in[7];
  float* out = (float*)d_out;

  bf16_t* Kt = (bf16_t*)d_ws;
  bf16_t* Qt = Kt + (size_t)BB * NN * CC;
  bf16_t* V  = Qt + (size_t)BB * NN * CC;
  float* colmax = (float*)(V + (size_t)BB * CC * NN);
  float* colsum = colmax + (size_t)BB * NN;

  proj_kernel<<<dim3(BB * (NN / K1_NB)), 256, 0, stream>>>(
      x, w1, b1, w2, b2, w3, b3, Kt, Qt, V);
  stats_kernel<<<dim3(BB * (NN / 128)), 256, 0, stream>>>(
      Qt, Kt, colmax, colsum);
  attnout_kernel<<<dim3(BB * (NN / 16)), 256, 0, stream>>>(
      x, Qt, Kt, V, colmax, colsum, al, out);
  (void)in_sizes; (void)n_in; (void)out_size; (void)ws_size;
}
